// SingleHeadAttention_5772436046449
// MI455X (gfx1250) — compile-verified, matrix operand order checked by reading, not run on hardware
//
#include <hip/hip_runtime.h>

#define HID 1024
#define HD 64
#define BSZ 4
#define SEQ 4096
#define ROWS (BSZ * SEQ) // 16384

typedef __attribute__((ext_vector_type(16))) __bf16 v16bf;
typedef __attribute__((ext_vector_type(8)))  float  v8f;
typedef __attribute__((__vector_size__(16))) int v4i_;
typedef __attribute__((address_space(1))) v4i_ gv4i; // global int4
typedef __attribute__((address_space(3))) v4i_ lv4i; // LDS int4

// ---------------- gfx1250 async LDS copy (guarded) ----------------
#if __has_builtin(__builtin_amdgcn_global_load_async_to_lds_b128) && \
    __has_builtin(__builtin_amdgcn_s_wait_asynccnt)
#define HAVE_ASYNC_LDS 1
#else
#define HAVE_ASYNC_LDS 0
#endif

// Copy 16 bytes global -> LDS for this lane.
static __device__ __forceinline__ void copy16_to_lds(__bf16* dst,
                                                     const __bf16* src) {
#if HAVE_ASYNC_LDS
  __builtin_amdgcn_global_load_async_to_lds_b128((gv4i*)src, (lv4i*)dst, 0, 0);
#else
  *(uint4*)dst = *(const uint4*)src;
#endif
}
static __device__ __forceinline__ void async_copy_wait() {
#if HAVE_ASYNC_LDS
  __builtin_amdgcn_s_wait_asynccnt(0);
#endif
}

// ---- WMMA helper: D = A(16x32 bf16) x B(32x16 bf16) + C(16x16 f32) ----
static __device__ __forceinline__ v8f wmma_bf16(v16bf a, v16bf b, v8f c) {
  return __builtin_amdgcn_wmma_f32_16x16x32_bf16(false, a, false, b, (short)0, c,
                                                 false, false);
}

// Load a 16-element bf16 A/B fragment slice for this lane.
// `row` points at the lane's matrix row, at the start of the 32-wide K window.
// kb = 8*(lane>>4). ISA layout: elements 0..7 = K kb..kb+7, 8..15 = K kb+16..kb+23.
static __device__ __forceinline__ v16bf load_frag_bf16(const __bf16* __restrict__ row,
                                                       int kb) {
  v16bf f;
#pragma unroll
  for (int e = 0; e < 8; ++e) {
    f[e]     = row[kb + e];
    f[e + 8] = row[kb + 16 + e];
  }
  return f;
}

// ---------------- Kernel 0: fp32 -> bf16 weight conversion ----------------
__global__ void cvt_f32_bf16(const float* __restrict__ s, __bf16* __restrict__ d,
                             int n) {
  int i = blockIdx.x * blockDim.x + threadIdx.x;
  if (i < n) d[i] = (__bf16)s[i];
}

// ---------------- Kernel 1: fused QKV projection (bf16 WMMA) ----------------
// Y[m, n] = sum_k x[m, k] * W[n, k];  M=16384, N=64, K=1024.
// Block = 8 waves: 2 M-tiles x 4 N-tiles. grid.y selects Q/K/V.
// V is stored TRANSPOSED: Vt[d * ROWS + row] so attention reads it contiguously.
__global__ void __launch_bounds__(256) qkv_proj_kernel(
    const float* __restrict__ x, const __bf16* __restrict__ Wq,
    const __bf16* __restrict__ Wk, const __bf16* __restrict__ Wv,
    __bf16* __restrict__ Q, __bf16* __restrict__ K, __bf16* __restrict__ Vt) {
  const int lane = threadIdx.x & 31, wave = threadIdx.x >> 5;
  const int l16 = lane & 15, half = lane >> 4, kb = half * 8;
  const int mt = wave >> 2, nt = wave & 3;
  const int m0 = blockIdx.x * 32 + mt * 16;
  const int n0 = nt * 16;
  const __bf16* W = (blockIdx.y == 0) ? Wq : (blockIdx.y == 1) ? Wk : Wv;

  const float*  arow = x + (size_t)(m0 + l16) * HID;
  const __bf16* wrow = W + (size_t)(n0 + l16) * HID;

  v8f acc = {};
#pragma unroll 4
  for (int k0 = 0; k0 < HID; k0 += 32) {
    v16bf af;
#pragma unroll
    for (int e = 0; e < 8; ++e) {
      af[e]     = (__bf16)arow[k0 + kb + e];
      af[e + 8] = (__bf16)arow[k0 + kb + 16 + e];
    }
    v16bf bf = load_frag_bf16(wrow + k0, kb);
    acc = wmma_bf16(af, bf, acc);
  }
  if (blockIdx.y == 2) {
    // V: transposed store Vt[d][row]; per lane the 8 rows are contiguous.
#pragma unroll
    for (int r = 0; r < 8; ++r)
      Vt[(size_t)(n0 + l16) * ROWS + m0 + r + 8 * half] = (__bf16)acc[r];
  } else {
    __bf16* Y = (blockIdx.y == 0) ? Q : K;
#pragma unroll
    for (int r = 0; r < 8; ++r)
      Y[(size_t)(m0 + r + 8 * half) * HD + n0 + l16] = (__bf16)acc[r];
  }
}

// ---------------- Kernel 2: causal flash attention (bf16 WMMA) ----------------
// Block = 8 waves = 128 query rows; each wave owns 16 rows.
// K/Vt tiles (32 keys) staged cooperatively into double-buffered LDS with
// async loads; per-wave online softmax; P transposed C->A layout through LDS.
__global__ void __launch_bounds__(256) attn_kernel(
    const __bf16* __restrict__ Q, const __bf16* __restrict__ Kb,
    const __bf16* __restrict__ Vt, __bf16* __restrict__ AO) {
  const int tid = threadIdx.x;
  const int lane = tid & 31, wave = tid >> 5;
  const int l16 = lane & 15, half = lane >> 4, kb = half * 8;
  const int batch = blockIdx.x >> 5; // 32 query-chunks of 128 per batch
  const int qbase = (blockIdx.x & 31) * 128;
  const int qloc = qbase + wave * 16;
  const size_t rowbase = (size_t)batch * SEQ;

  __shared__ __bf16 lK[2][32 * HD];  // key tile, row-major [32 keys][64 d]
  __shared__ __bf16 lV[2][HD * 32];  // Vt tile, row-major [64 d][32 keys]
  __shared__ __bf16 pl[8][16 * 32];  // per-wave P transpose buffer

  // Q fragments: 16x64 = two 16x32 A-fragments (direct from global)
  v16bf qf[2];
#pragma unroll
  for (int c = 0; c < 2; ++c)
    qf[c] = load_frag_bf16(Q + (rowbase + qloc + l16) * HD + 32 * c, kb);

  float m_i[8], l_i[8];
  v8f o[4];
#pragma unroll
  for (int r = 0; r < 8; ++r) { m_i[r] = -1e30f; l_i[r] = 0.0f; }
#pragma unroll
  for (int n = 0; n < 4; ++n) o[n] = (v8f){};

  const int nkb_w   = (qloc + 16 + 31) >> 5;   // this wave's causal key blocks
  const int nkb_blk = (qbase + 128 + 31) >> 5; // block-uniform trip count

  // Cooperative stage of one 32-key tile (K: 4KB, Vt: 4KB) into buffer `buf`.
  auto stage = [&](int jb, int buf) {
    const int k0 = jb * 32;
    copy16_to_lds(&lK[buf][(tid >> 3) * HD + (tid & 7) * 8],
                  Kb + (rowbase + k0 + (tid >> 3)) * HD + (tid & 7) * 8);
    copy16_to_lds(&lV[buf][(tid >> 2) * 32 + (tid & 3) * 8],
                  Vt + (size_t)(tid >> 2) * ROWS + rowbase + k0 + (tid & 3) * 8);
  };

  stage(0, 0);
  async_copy_wait();
  __syncthreads();

  for (int jb = 0; jb < nkb_blk; ++jb) {
    const int buf = jb & 1;
    const int k0 = jb * 32;
    if (jb + 1 < nkb_blk) stage(jb + 1, buf ^ 1); // prefetch next tile (async)

    if (jb < nkb_w) {
      // ---- scores: two 16x16 C tiles (keys k0..+15 and k0+16..+31) ----
      v8f s0 = {}, s1 = {};
#pragma unroll
      for (int c = 0; c < 2; ++c) {
        v16bf kf0 = load_frag_bf16(&lK[buf][l16 * HD + 32 * c], kb);
        s0 = wmma_bf16(qf[c], kf0, s0);
        v16bf kf1 = load_frag_bf16(&lK[buf][(16 + l16) * HD + 32 * c], kb);
        s1 = wmma_bf16(qf[c], kf1, s1);
      }
      // ---- online softmax update, per row r (+8*half) ----
      const float scale = 0.125f; // 1/sqrt(64)
#pragma unroll
      for (int r = 0; r < 8; ++r) {
        const int q_idx = qloc + r + 8 * half;
        float v0 = s0[r] * scale, v1 = s1[r] * scale;
        if (k0 + l16 > q_idx)      v0 = -1e30f; // causal mask (finite)
        if (k0 + 16 + l16 > q_idx) v1 = -1e30f;
        float rm = fmaxf(v0, v1);
#pragma unroll
        for (int off = 8; off >= 1; off >>= 1)
          rm = fmaxf(rm, __shfl_xor(rm, off, 16));
        const float mnew = fmaxf(m_i[r], rm);
        const float corr = __expf(m_i[r] - mnew);
        v0 = __expf(v0 - mnew);
        v1 = __expf(v1 - mnew);
        float rs = v0 + v1;
#pragma unroll
        for (int off = 8; off >= 1; off >>= 1) rs += __shfl_xor(rs, off, 16);
        l_i[r] = l_i[r] * corr + rs;
        m_i[r] = mnew;
#pragma unroll
        for (int n = 0; n < 4; ++n) o[n][r] *= corr;
        const int prow = r + 8 * half;
        pl[wave][prow * 32 + l16]      = (__bf16)v0;
        pl[wave][prow * 32 + 16 + l16] = (__bf16)v1;
      }
      // in-wave DS ordering covers the RAW; read back as A-fragment
      v16bf pf = load_frag_bf16(&pl[wave][l16 * 32], kb);

      // ---- O += P(16x32) x V(32x64): 4 WMMAs from the Vt LDS tile ----
#pragma unroll
      for (int n = 0; n < 4; ++n) {
        v16bf vf = load_frag_bf16(&lV[buf][(n * 16 + l16) * 32], kb);
        o[n] = wmma_bf16(pf, vf, o[n]);
      }
    }

    async_copy_wait(); // my prefetch landed
    __syncthreads();   // everyone done reading buf / writing buf^1
  }

  // ---- epilogue: O / l, store bf16 ----
#pragma unroll
  for (int r = 0; r < 8; ++r) {
    const float inv = 1.0f / l_i[r];
    const size_t orow = (rowbase + qloc + r + 8 * half) * HD;
#pragma unroll
    for (int n = 0; n < 4; ++n)
      AO[orow + n * 16 + l16] = (__bf16)(o[n][r] * inv);
  }
}

// ---------------- Kernel 3: output projection (bf16 WMMA, f32 out) ----------------
// out[m, n] = sum_d AO[m, d] * Wo[n, d];  M=16384, N=1024, K=64.
__global__ void __launch_bounds__(256) out_proj_kernel(
    const __bf16* __restrict__ AO, const __bf16* __restrict__ Wo,
    float* __restrict__ out) {
  const int lane = threadIdx.x & 31, wave = threadIdx.x >> 5;
  const int l16 = lane & 15, half = lane >> 4, kb = half * 8;
  const int m0 = blockIdx.x * 16;
  const int n0 = (blockIdx.y * 8 + wave) * 16;
  v8f acc = {};
#pragma unroll
  for (int c = 0; c < 2; ++c) {
    v16bf af = load_frag_bf16(AO + (size_t)(m0 + l16) * HD + 32 * c, kb);
    v16bf bf = load_frag_bf16(Wo + (size_t)(n0 + l16) * HD + 32 * c, kb);
    acc = wmma_bf16(af, bf, acc);
  }
#pragma unroll
  for (int r = 0; r < 8; ++r)
    out[(size_t)(m0 + r + 8 * half) * HID + n0 + l16] = acc[r];
}

extern "C" void kernel_launch(void* const* d_in, const int* in_sizes, int n_in,
                              void* d_out, int out_size, void* d_ws, size_t ws_size,
                              hipStream_t stream) {
  const float* x  = (const float*)d_in[0];
  const float* Wq = (const float*)d_in[1];
  const float* Wk = (const float*)d_in[2];
  const float* Wv = (const float*)d_in[3];
  const float* Wo = (const float*)d_in[4];
  float* out = (float*)d_out;

  // workspace layout (bf16): 4 weight copies (512 KB) + Q/K/Vt/AO (8 MB)
  __bf16* Wq_bf = (__bf16*)d_ws;
  __bf16* Wk_bf = Wq_bf + (size_t)HD * HID;
  __bf16* Wv_bf = Wk_bf + (size_t)HD * HID;
  __bf16* Wo_bf = Wv_bf + (size_t)HD * HID;
  __bf16* Qb    = Wo_bf + (size_t)HID * HD;
  __bf16* Kb    = Qb + (size_t)ROWS * HD;
  __bf16* Vtb   = Kb + (size_t)ROWS * HD;
  __bf16* AOb   = Vtb + (size_t)ROWS * HD;

  const int wn = HD * HID;
  cvt_f32_bf16<<<(wn + 255) / 256, 256, 0, stream>>>(Wq, Wq_bf, wn);
  cvt_f32_bf16<<<(wn + 255) / 256, 256, 0, stream>>>(Wk, Wk_bf, wn);
  cvt_f32_bf16<<<(wn + 255) / 256, 256, 0, stream>>>(Wv, Wv_bf, wn);
  cvt_f32_bf16<<<(wn + 255) / 256, 256, 0, stream>>>(Wo, Wo_bf, wn);

  qkv_proj_kernel<<<dim3(ROWS / 32, 3), 256, 0, stream>>>(x, Wq_bf, Wk_bf, Wv_bf,
                                                          Qb, Kb, Vtb);
  attn_kernel<<<ROWS / 128, 256, 0, stream>>>(Qb, Kb, Vtb, AOb);
  out_proj_kernel<<<dim3(ROWS / 16, HID / 128), 256, 0, stream>>>(AOb, Wo_bf, out);
}
